// QueryNetwork_13769665151413
// MI455X (gfx1250) — matrix operand order repaired and checked by reading, not compiled, not run on hardware
//
#include <hip/hip_runtime.h>

typedef __attribute__((ext_vector_type(16))) _Float16 v16h;
typedef __attribute__((ext_vector_type(8)))  float    v8f;

#define N_NODES   585
#define MEM_DIM   256
#define BEAM      32
#define CFAN      8
#define POOL_MAX  288   // 32 old + 256 children
#define NEG_SEL   (-1e9f)
#define NEG_FINAL (-99999.0f)

#if __has_builtin(__builtin_amdgcn_global_load_async_to_lds_b32) && \
    __has_builtin(__builtin_amdgcn_s_wait_asynccnt)
#define USE_ASYNC_LDS 1
#else
#define USE_ASYNC_LDS 0
#endif

// Map float to an order-preserving unsigned key (ascending).
__device__ __forceinline__ unsigned fkey(float f) {
  unsigned u = __float_as_uint(f);
  return (u & 0x80000000u) ? ~u : (u | 0x80000000u);
}

// One wave computes dot(q, row) with coalesced 32-lane loads + shuffle reduce.
__device__ __forceinline__ float wave_dot(const float* __restrict__ row,
                                          const float* __restrict__ qs, int lane) {
  float acc = 0.f;
#pragma unroll
  for (int d = 0; d < MEM_DIM / 32; ++d)
    acc += row[lane + d * 32] * qs[lane + d * 32];
#pragma unroll
  for (int off = 16; off > 0; off >>= 1) acc += __shfl_down(acc, off, 32);
  return __shfl(acc, 0, 32);
}

// Stable top-32 selection over pool[0..P-1]; rebuilds beam arrays.
// Key = ord(score)<<32 | ~poolIdx  => max key = highest score, ties -> lowest idx.
__device__ __forceinline__ void topk_rebuild(
    int P, int t, int lane,
    const float* pscore, const int* pidx,
    const unsigned char* pvalid, const unsigned char* pfin,
    unsigned long long* keys,
    float* bscore, int* bidx, unsigned char* bvalid, unsigned char* bfin) {
  for (int j = t; j < P; j += 256)
    keys[j] = ((unsigned long long)fkey(pscore[j]) << 32) | (unsigned)(~j);
  __syncthreads();
  if (t < 32) {
    unsigned long long kreg[9];
#pragma unroll
    for (int r = 0; r < 9; ++r) {
      int j = lane + r * 32;
      kreg[r] = (j < P) ? keys[j] : 0ull;
    }
    int myj = 0;
    for (int s = 0; s < 32; ++s) {
      unsigned long long m = 0ull;
#pragma unroll
      for (int r = 0; r < 9; ++r) m = (kreg[r] > m) ? kreg[r] : m;
#pragma unroll
      for (int off = 16; off > 0; off >>= 1) {
        unsigned long long o = __shfl_down(m, off, 32);
        m = (o > m) ? o : m;
      }
      m = __shfl(m, 0, 32);
      int j = (int)(~(unsigned)(m & 0xFFFFFFFFull));
#pragma unroll
      for (int r = 0; r < 9; ++r)
        if ((j >> 5) == r && (j & 31) == lane) kreg[r] = 0ull;  // consume
      if (lane == s) myj = j;
    }
    bscore[lane] = pscore[myj];
    bidx[lane]   = pidx[myj];
    bvalid[lane] = pvalid[myj];
    bfin[lane]   = pfin[myj];
  }
  __syncthreads();
}

__global__ void __launch_bounds__(256)
beam_kernel(const float* __restrict__ tree_mem, const float* __restrict__ query,
            float* __restrict__ out_resp, float* __restrict__ out_att,
            float* __restrict__ X) {
  __shared__ float qs[MEM_DIM];
  __shared__ float pscore[POOL_MAX];
  __shared__ int pidx[POOL_MAX];
  __shared__ unsigned char pvalid[POOL_MAX];
  __shared__ unsigned char pfin[POOL_MAX];
  __shared__ unsigned long long keys[POOL_MAX];
  __shared__ float bscore[BEAM];
  __shared__ int bidx[BEAM];
  __shared__ unsigned char bvalid[BEAM];
  __shared__ unsigned char bfin[BEAM];
  __shared__ float att[BEAM];
  __shared__ int rows[BEAM];

  const int t = threadIdx.x, lane = t & 31, wv = t >> 5;
  const int b = blockIdx.x;
  const float* memb = tree_mem + (size_t)b * N_NODES * MEM_DIM;

  qs[t] = query[(size_t)b * MEM_DIM + t];
  __syncthreads();

  // ---- initial beam: candidates k=0..31, node=k+1, valid = k < C
  for (int i = 0; i < 4; ++i) {
    int k = wv * 4 + i;
    int node = k + 1;
    bool v = (k < CFAN);
    float sc = NEG_SEL;
    if (v) sc = wave_dot(memb + (size_t)node * MEM_DIM, qs, lane);
    if (lane == 0) { pscore[k] = sc; pidx[k] = node; pvalid[k] = v ? 1 : 0; pfin[k] = 0; }
  }
  __syncthreads();
  topk_rebuild(32, t, lane, pscore, pidx, pvalid, pfin, keys,
               bscore, bidx, bvalid, bfin);

  // ---- 3 beam-expansion levels
  for (int lvl = 0; lvl < 3; ++lvl) {
    if (t < 32) { pscore[t] = bscore[t]; pidx[t] = bidx[t]; pvalid[t] = bvalid[t]; pfin[t] = 1; }
    __syncthreads();
    for (int i = 0; i < 32; ++i) {                 // 8 waves x 32 = 256 children
      int j = wv * 32 + i;
      int k = j >> 3, c = j & 7;
      int node = bidx[k] * CFAN + 1 + c;
      bool cv = bvalid[k] && !bfin[k] && (node < N_NODES);  // wave-uniform
      float sc = NEG_SEL;
      if (cv) sc = wave_dot(memb + (size_t)node * MEM_DIM, qs, lane);
      if (lane == 0) {
        int p = 32 + j;
        pscore[p] = sc; pidx[p] = node; pvalid[p] = cv ? 1 : 0; pfin[p] = 0;
      }
    }
    __syncthreads();
    topk_rebuild(POOL_MAX, t, lane, pscore, pidx, pvalid, pfin, keys,
                 bscore, bidx, bvalid, bfin);
  }

  // ---- final attention: recompute scores on clipped indices
  for (int i = 0; i < 4; ++i) {
    int k = wv * 4 + i;
    int node = bidx[k];
    node = node < 0 ? 0 : (node >= N_NODES ? N_NODES - 1 : node);
    float sc = wave_dot(memb + (size_t)node * MEM_DIM, qs, lane);
    if (lane == 0) { att[k] = sc; rows[k] = node; }
  }
  __syncthreads();
  if (t < 32) {
    float v = bvalid[t] ? att[t] : NEG_FINAL;
    float m = v;
#pragma unroll
    for (int off = 16; off > 0; off >>= 1) {
      float o = __shfl_down(m, off, 32);
      m = (o > m) ? o : m;
    }
    m = __shfl(m, 0, 32);
    float e = __expf(v - m);
    float s = e;
#pragma unroll
    for (int off = 16; off > 0; off >>= 1) s += __shfl_down(s, off, 32);
    s = __shfl(s, 0, 32);
    float a = e / s;
    att[t] = a;
    out_att[(size_t)b * BEAM + t] = a;
  }
  __syncthreads();

  // ---- response[d] = sum_k att[k] * mem[rows[k]][d]   (coalesced over d = t)
  float acc = 0.f;
#pragma unroll 4
  for (int k = 0; k < BEAM; ++k)
    acc += att[k] * memb[(size_t)rows[k] * MEM_DIM + t];
  out_resp[(size_t)b * MEM_DIM + t] = acc;
  X[(size_t)b * 512 + t] = acc;          // h = [response | query]
  X[(size_t)b * 512 + MEM_DIM + t] = qs[t];
}

// GEMM: C[M=1024,N] = relu?( A[M,K] @ B[K,N] + bias[N] ), fp32 in/out, f16 WMMA core.
// Compile-time K,N -> fully unrolled K loop (16x/8x static v_wmma).
// B tiles staged global->LDS with GLOBAL_LOAD_ASYNC_TO_LDS_B32 (ASYNCcnt),
// double-buffered; drain with s_wait_asynccnt 0 before the workgroup barrier.
template <int N, int K, int RELU>
__global__ void __launch_bounds__(256)
gemm_relu_wmma(const float* __restrict__ A, const float* __restrict__ B,
               const float* __restrict__ bias, float* __restrict__ C) {
  __shared__ float Bs[2][32 * 16];   // K-slab of B (f32), shared by all 8 waves

  const int t = threadIdx.x, lane = t & 31, wv = t >> 5;
  const int rowBase = blockIdx.y * 128 + wv * 16;
  const int colBase = blockIdx.x * 16;

  // Stage B tile (32K x 16N) for k0=kk into Bs[buf].
  auto stage = [&](int kk, int buf) {
#pragma unroll
    for (int i = 0; i < 2; ++i) {
      int e = t + i * 256;
      int k = e >> 4, n = e & 15, gn = colBase + n;
      if (gn < N) {
#if USE_ASYNC_LDS
        auto g = reinterpret_cast<int __attribute__((address_space(1)))*>(
            reinterpret_cast<uintptr_t>(&B[(size_t)(kk + k) * N + gn]));
        auto l = reinterpret_cast<int __attribute__((address_space(3)))*>(
            reinterpret_cast<uintptr_t>(&Bs[buf][e]));
        __builtin_amdgcn_global_load_async_to_lds_b32(g, l, 0, 0);
#else
        Bs[buf][e] = B[(size_t)(kk + k) * N + gn];
#endif
      }
#if !USE_ASYNC_LDS
      else Bs[buf][e] = 0.f;
#endif
    }
  };

  v8f c = {};
  stage(0, 0);

  constexpr int NT = K / 32;
#pragma unroll
  for (int ti = 0; ti < NT; ++ti) {
    const int kk = ti * 32;
    const int buf = ti & 1;
#if USE_ASYNC_LDS
    __builtin_amdgcn_s_wait_asynccnt(0);   // own async copies into buf done
#endif
    __syncthreads();                       // everyone's copies done / reads drained
    if (ti + 1 < NT) stage(kk + 32, buf ^ 1);

    // A fragment (ISA 16-bit 16x32 layout): lane<16 holds K {0..7,16..23},
    // lane>=16 holds K {8..15,24..31}, row M = lane&15.
    const int row = rowBase + (lane & 15);
    const float* ap = A + (size_t)row * K + kk + ((lane & 16) ? 8 : 0);
    if (ti + 1 < NT) __builtin_prefetch(ap + 32, 0, 0);
    v16h a;
#pragma unroll
    for (int j = 0; j < 8; ++j) {
      a[j]     = (_Float16)ap[j];
      a[j + 8] = (_Float16)ap[16 + j];
    }
    // B fragment: lane n = lane&15 holds column n; lanes 0-15 K {0..7,16..23}, 16-31 K {8..15,24..31}.
    v16h bf;
    const int n = lane & 15, kof = (lane & 16) ? 8 : 0;
#pragma unroll
    for (int j = 0; j < 16; ++j) bf[j] = (_Float16)Bs[buf][(kof + (j & 7) + ((j & 8) << 1)) * 16 + n];

    c = __builtin_amdgcn_wmma_f32_16x16x32_f16(
        /*neg_a=*/false, a, /*neg_b=*/false, bf,
        /*c_mod=*/(short)0, c, /*reuse_a=*/false, /*reuse_b=*/false);
  }

  // Epilogue per ISA C/D layout: VGPR r -> M = r + 8*(lane>=16), N = lane&15.
  const int n = lane & 15, gn = colBase + n;
  const int mbase = rowBase + ((lane & 16) ? 8 : 0);
  if (gn < N) {
    float bv = bias[gn];
#pragma unroll
    for (int r = 0; r < 8; ++r) {
      float v = c[r] + bv;
      if (RELU) v = fmaxf(v, 0.f);
      C[(size_t)(mbase + r) * N + gn] = v;
    }
  }
}

extern "C" void kernel_launch(void* const* d_in, const int* in_sizes, int n_in,
                              void* d_out, int out_size, void* d_ws, size_t ws_size,
                              hipStream_t stream) {
  (void)in_sizes; (void)n_in; (void)out_size; (void)ws_size;
  const float* tree_mem = (const float*)d_in[0];
  const float* query    = (const float*)d_in[1];
  const float* W1 = (const float*)d_in[2];
  const float* b1 = (const float*)d_in[3];
  const float* W2 = (const float*)d_in[4];
  const float* b2 = (const float*)d_in[5];
  const float* W3 = (const float*)d_in[6];
  const float* b3 = (const float*)d_in[7];

  const int BS = 1024;
  float* out      = (float*)d_out;            // [1024, 3]
  float* out_resp = out + BS * 3;             // [1024, 256]
  float* out_att  = out_resp + BS * 256;      // [1024, 32]

  float* X  = (float*)d_ws;                   // [1024, 512]
  float* H1 = X + BS * 512;                   // [1024, 256]
  float* H2 = H1 + BS * 256;                  // [1024, 512]

  beam_kernel<<<BS, 256, 0, stream>>>(tree_mem, query, out_resp, out_att, X);
  gemm_relu_wmma<256, 512, 1><<<dim3(16, 8), 256, 0, stream>>>(X,  W1, b1, H1);
  gemm_relu_wmma<512, 256, 1><<<dim3(32, 8), 256, 0, stream>>>(H1, W2, b2, H2);
  gemm_relu_wmma<  3, 512, 0><<<dim3( 1, 8), 256, 0, stream>>>(H2, W3, b3, out);
}
